// IncrementalVGG16V1_10196252361453
// MI455X (gfx1250) — matrix operand order repaired and checked by reading, not compiled, not run on hardware
//
#include <hip/hip_runtime.h>
#include <hip/hip_bf16.h>
#include <stdint.h>

// ---------------------------------------------------------------------------
// Incremental VGG16 for MI455X (gfx1250, wave32, WMMA).
//  - full-image forward: exact fp32 VALU (batch-1, ~15 GFLOP, seeds caches)
//  - incremental patch convs (~1.15 TFLOP) + FC layers: bf16 WMMA
//    (v_wmma_f32_16x16x32_bf16, fp32 accumulate)
//  - patch tensors are NHWC and weights are packed with k=(ky*3+kx)*C+c, so an
//    A fragment is two contiguous global_load_b128 and a B fragment is one 32B
//    load of the pre-packed fragment layout: the conv inner loop is
//    1 ds_load_b64 + 4 b128(A) + 4x32B(B) + 8 WMMA per 32-wide K step.
// ---------------------------------------------------------------------------

#define PSTRIDE 131072  // per-batch patch buffer stride (max 512*16*16 elems)

typedef __attribute__((ext_vector_type(16))) __bf16        v16bf;
typedef __attribute__((ext_vector_type(8)))  float         v8f;
typedef __attribute__((ext_vector_type(8)))  unsigned int  v8u;
typedef __attribute__((ext_vector_type(4)))  unsigned int  v4u;

static __device__ __forceinline__ unsigned short f2bf(float f) {
    unsigned int x = __builtin_bit_cast(unsigned int, f);
    x += 0x7fffu + ((x >> 16) & 1u);            // round-to-nearest-even
    return (unsigned short)(x >> 16);
}
static __device__ __forceinline__ float bf2f(unsigned short h) {
    unsigned int u = ((unsigned int)h) << 16;
    return __builtin_bit_cast(float, u);
}
static __device__ __forceinline__ unsigned int pk2(unsigned short lo, unsigned short hi) {
    return (unsigned int)lo | ((unsigned int)hi << 16);
}

// ------------------------- full-image forward (fp32) ------------------------

__global__ void conv_full(const float* __restrict__ in, const float* __restrict__ w,
                          const float* __restrict__ bias, float* __restrict__ out,
                          int C, int H, int O) {
    long long idx = (long long)blockIdx.x * blockDim.x + threadIdx.x;
    long long tot = (long long)O * H * H;
    if (idx >= tot) return;
    int x = (int)(idx % H);
    int y = (int)((idx / H) % H);
    int o = (int)(idx / ((long long)H * H));
    float s = bias[o];
    const float* wp = w + (size_t)o * C * 9;
    for (int c = 0; c < C; ++c) {
        const float* ip = in + (size_t)c * H * H;
        const float* wc = wp + c * 9;
#pragma unroll
        for (int ky = 0; ky < 3; ++ky) {
            int yy = y + ky - 1;
            if (yy < 0 || yy >= H) continue;
#pragma unroll
            for (int kx = 0; kx < 3; ++kx) {
                int xx = x + kx - 1;
                if (xx < 0 || xx >= H) continue;
                s += ip[(size_t)yy * H + xx] * wc[ky * 3 + kx];
            }
        }
    }
    out[idx] = s > 0.f ? s : 0.f;
}

__global__ void pool_full(const float* __restrict__ in, float* __restrict__ out,
                          int C, int H, int Ho) {
    long long idx = (long long)blockIdx.x * blockDim.x + threadIdx.x;
    long long tot = (long long)C * Ho * Ho;
    if (idx >= tot) return;
    int x = (int)(idx % Ho);
    int y = (int)((idx / Ho) % Ho);
    int c = (int)(idx / ((long long)Ho * Ho));
    const float* ip = in + (size_t)c * H * H;
    float a = ip[(size_t)(2 * y) * H + 2 * x];
    float b = ip[(size_t)(2 * y) * H + 2 * x + 1];
    float d = ip[(size_t)(2 * y + 1) * H + 2 * x];
    float e = ip[(size_t)(2 * y + 1) * H + 2 * x + 1];
    out[idx] = fmaxf(fmaxf(a, b), fmaxf(d, e));
}

// ----------------------- patch pipeline (NHWC bf16) -------------------------

__global__ void init_patch(const float* __restrict__ patch, unsigned short* __restrict__ P) {
    long long idx = (long long)blockIdx.x * blockDim.x + threadIdx.x;
    if (idx >= 256LL * 3 * 16 * 16) return;
    int pix = (int)(idx % (3 * 256));      // (y*16+x)*3 + c
    int b   = (int)(idx / (3 * 256));
    int c   = pix % 3;
    int yx  = pix / 3;
    P[(size_t)b * PSTRIDE + pix] = f2bf(patch[c * 256 + yx]);
}

// ol = clamp(ceil((p + li - k + 1)/s), 0, size_out - out_p)  (reference semantics)
__global__ void update_locs(const int* __restrict__ li, int* __restrict__ lo,
                            int s, int p, int k, int out_p, int sizeOut) {
    int b = threadIdx.x;
    if (b >= 256) return;
#pragma unroll
    for (int d = 0; d < 2; ++d) {
        int a  = p + li[2 * b + d] - k + 1;
        int ol = (a > 0) ? (a + s - 1) / s : 0;
        if (ol + out_p > sizeOut) ol = sizeOut - out_p;
        if (ol < 0) ol = 0;
        lo[2 * b + d] = ol;
    }
}

// Build X[b][in_p][in_p][C] (bf16 NHWC): padded window of the cached NCHW fp32
// activation, interior overwritten by previous NHWC patch output.
__global__ void assemble_patch(const float* __restrict__ act, int sizeIn,
                               const unsigned short* __restrict__ Pin, int ps,
                               const int* __restrict__ li, const int* __restrict__ lo,
                               unsigned short* __restrict__ X,
                               int C, int in_p, int s, int p, long long tot) {
    long long idx = (long long)blockIdx.x * blockDim.x + threadIdx.x;
    if (idx >= tot) return;
    int c = (int)(idx % C);
    long long t = idx / C;
    int j = (int)(t % in_p); t /= in_p;
    int i = (int)(t % in_p);
    int b = (int)(t / in_p);
    int u0 = s * lo[2 * b]     - p;
    int u1 = s * lo[2 * b + 1] - p;
    int rx = li[2 * b]     - u0;
    int ry = li[2 * b + 1] - u1;
    int rmax = in_p - ps;
    rx = rx < 0 ? 0 : (rx > rmax ? rmax : rx);
    ry = ry < 0 ? 0 : (ry > rmax ? rmax : ry);
    unsigned short val;
    if (i >= rx && i < rx + ps && j >= ry && j < ry + ps) {
        val = Pin[(size_t)b * PSTRIDE + (size_t)((i - rx) * ps + (j - ry)) * C + c];
    } else {
        int a0 = u0 + i, a1 = u1 + j;
        val = (a0 >= 0 && a0 < sizeIn && a1 >= 0 && a1 < sizeIn)
                  ? f2bf(act[(size_t)c * sizeIn * sizeIn + (size_t)a0 * sizeIn + a1])
                  : (unsigned short)0;
    }
    X[(size_t)b * PSTRIDE + (size_t)(i * in_p + j) * C + c] = val;
}

__global__ void pool_patch(const unsigned short* __restrict__ X, unsigned short* __restrict__ P,
                           int C, int in_p, int out_p, long long tot) {
    long long idx = (long long)blockIdx.x * blockDim.x + threadIdx.x;
    if (idx >= tot) return;
    int c = (int)(idx % C);
    long long t = idx / C;
    int x = (int)(t % out_p); t /= out_p;
    int y = (int)(t % out_p);
    int b = (int)(t / out_p);
    const unsigned short* ip = X + (size_t)b * PSTRIDE;
    float a0 = bf2f(ip[(size_t)((2 * y) * in_p + 2 * x) * C + c]);
    float a1 = bf2f(ip[(size_t)((2 * y) * in_p + 2 * x + 1) * C + c]);
    float a2 = bf2f(ip[(size_t)((2 * y + 1) * in_p + 2 * x) * C + c]);
    float a3 = bf2f(ip[(size_t)((2 * y + 1) * in_p + 2 * x + 1) * C + c]);
    float m  = fmaxf(fmaxf(a0, a1), fmaxf(a2, a3));
    P[(size_t)b * PSTRIDE + (size_t)(y * out_p + x) * C + c] = f2bf(m);
}

// Final NHWC patches -> dense NCHW [256][512*7*7] bf16 (reference reshape order)
__global__ void pool5_transpose(const unsigned short* __restrict__ in,
                                unsigned short* __restrict__ out) {
    long long idx = (long long)blockIdx.x * blockDim.x + threadIdx.x;
    if (idx >= 256LL * 25088) return;
    int k = (int)(idx % 25088);            // c*49 + (y*7+x)
    int b = (int)(idx / 25088);
    int c  = k / 49;
    int yx = k % 49;
    out[idx] = in[(size_t)b * PSTRIDE + (size_t)yx * 512 + c];
}

// -------------------- weight pre-pack (fragment layout) ---------------------
// Wpk layout: [n16][kt][lane][8 dwords]; lane lq=lane&15 -> n, g=lane>>4,
// dword v -> (k = kt*32 + 8g + 2v if v<4 else kt*32 + 16 + 8g + 2(v-4), k+1).  Cremap>0 reorders logical k to
// (ky*3+kx)*C + c (channel-innermost, matching NHWC A tensors): src = c*9 + r.
__global__ void pack_weights(const float* __restrict__ W, unsigned int* __restrict__ Wpk,
                             int K, int N, int KT, int Cremap, long long tot) {
    long long idx = (long long)blockIdx.x * blockDim.x + threadIdx.x;
    if (idx >= tot) return;
    int v    = (int)(idx & 7);
    int lane = (int)((idx >> 3) & 31);
    long long t = idx >> 8;
    int kt  = (int)(t % KT);
    int n16 = (int)(t / KT);
    int lq = lane & 15, g = lane >> 4;
    int n  = n16 * 16 + lq;
    int k0 = kt * 32 + ((v < 4) ? (8 * g + 2 * v) : (16 + 8 * g + 2 * (v - 4)));
    unsigned short b0 = 0, b1 = 0;
    if (n < N) {
        const float* Wr = W + (size_t)n * K;
        if (k0 < K) {
            int sk = Cremap ? ((k0 % Cremap) * 9 + k0 / Cremap) : k0;
            b0 = f2bf(Wr[sk]);
        }
        if (k0 + 1 < K) {
            int k1 = k0 + 1;
            int sk = Cremap ? ((k1 % Cremap) * 9 + k1 / Cremap) : k1;
            b1 = f2bf(Wr[sk]);
        }
    }
    Wpk[idx] = pk2(b0, b1);
}

// -------------------- WMMA implicit-GEMM VALID 3x3 conv ---------------------
// One wave: 32(M) x 64(N) tile (2 M-tiles x 4 N-subtiles, 8 WMMA/kstep).
// NHWC X + channel-innermost K: each A fragment half (8 consecutive k) is one
// aligned b128 load; the two byte-prescaled run offsets a lane needs per
// K-step sit adjacent in LDS (one ds_load_b64).
template <bool TAIL>
__global__ __launch_bounds__(256)
void conv_patch_wmma(const unsigned short* __restrict__ X,
                     const unsigned int* __restrict__ Wpk,
                     const float* __restrict__ bias,
                     unsigned short* __restrict__ Pout,
                     int C, int in_p, int out_p, int N, int KT) {
    __shared__ unsigned tbl[640];                  // 4*KT <= 576 (TAIL: K<=27)
    int K = C * 9;
    if (TAIL) {
        for (int k = threadIdx.x; k < K; k += 256) {
            int r = k / C, c = k % C;
            tbl[k] = (unsigned)(((r / 3) * in_p + (r % 3)) * C + c);   // elem offset
        }
    } else {
        int T2 = KT * 2;
        for (int t2 = threadIdx.x; t2 < T2; t2 += 256) {
            int kt = t2 >> 1, g = t2 & 1;
            int k0 = (kt * 4 + g) * 8;             // run k = kt*32 + 8g
            int k1 = (kt * 4 + 2 + g) * 8;         // run k = kt*32 + 16 + 8g
            int r0 = k0 / C, c0 = k0 % C;
            int r1 = k1 / C, c1 = k1 % C;
            tbl[t2 * 2]     = (unsigned)(((((r0 / 3) * in_p + (r0 % 3)) * C) + c0) * 2);  // bytes
            tbl[t2 * 2 + 1] = (unsigned)(((((r1 / 3) * in_p + (r1 % 3)) * C) + c1) * 2);
        }
    }
    __syncthreads();

    int wid  = (int)((blockIdx.x * blockDim.x + threadIdx.x) >> 5);
    int lane = threadIdx.x & 31;
    int op2  = out_p * out_p;
    int Mt2  = (256 * op2) >> 5;                   // pairs of 16-row tiles
    int n64  = N >> 6;
    if (wid >= Mt2 * n64) return;                  // wave-uniform exit
    int mt2 = wid % Mt2, nt = wid / Mt2;
    int g = lane >> 4, lq = lane & 15;

    unsigned rb[2];
    const char* XbB[2];
#pragma unroll
    for (int mi = 0; mi < 2; ++mi) {
        int m  = (mt2 * 2 + mi) * 16 + lq;
        int b  = m / op2;
        int r0 = m % op2;
        int oy = r0 / out_p, ox = r0 % out_p;
        rb[mi]  = (unsigned)b * PSTRIDE + (unsigned)(oy * in_p + ox) * C;
        XbB[mi] = (const char*)(X + rb[mi]);
    }
    const unsigned int* Bp[4];
#pragma unroll
    for (int sub = 0; sub < 4; ++sub) {
        int n16 = nt * 4 + sub;
        Bp[sub] = Wpk + (size_t)n16 * KT * 256 + lane * 8;
    }

    v8f acc[2][4];
#pragma unroll
    for (int mi = 0; mi < 2; ++mi)
#pragma unroll
        for (int s2 = 0; s2 < 4; ++s2)
#pragma unroll
            for (int e = 0; e < 8; ++e) acc[mi][s2][e] = 0.f;

    union AFrag { v4u q[2]; v8u u; v16bf bf; };

    for (int kt = 0; kt < KT; ++kt) {
        v16bf a[2];
        if (!TAIL) {
            uint2 o = *(const uint2*)&tbl[(kt * 2 + g) * 2];   // ds_load_b64
#pragma unroll
            for (int mi = 0; mi < 2; ++mi) {
                AFrag af;
                af.q[0] = *(const v4u*)(XbB[mi] + o.x);
                af.q[1] = *(const v4u*)(XbB[mi] + o.y);
                a[mi] = af.bf;
            }
        } else {
            int kb = kt * 32;
#pragma unroll
            for (int mi = 0; mi < 2; ++mi) {
                v8u au;
#pragma unroll
                for (int v = 0; v < 8; ++v) {
                    int k0 = kb + ((v < 4) ? (2 * v + 8 * g) : (16 + 2 * (v - 4) + 8 * g));
                    unsigned short e0 = (k0     < K) ? X[rb[mi] + tbl[k0]]     : (unsigned short)0;
                    unsigned short e1 = (k0 + 1 < K) ? X[rb[mi] + tbl[k0 + 1]] : (unsigned short)0;
                    au[v] = pk2(e0, e1);
                }
                a[mi] = __builtin_bit_cast(v16bf, au);
            }
        }
#pragma unroll
        for (int sub = 0; sub < 4; ++sub) {
            v8u bu = *(const v8u*)(Bp[sub] + (size_t)kt * 256);
            v16bf bfrag = __builtin_bit_cast(v16bf, bu);
#pragma unroll
            for (int mi = 0; mi < 2; ++mi)
                acc[mi][sub] = __builtin_amdgcn_wmma_f32_16x16x32_bf16(
                    false, a[mi], false, bfrag, (short)0, acc[mi][sub], false, false);
        }
    }

#pragma unroll
    for (int mi = 0; mi < 2; ++mi) {
#pragma unroll
        for (int sub = 0; sub < 4; ++sub) {
            int n = nt * 64 + sub * 16 + lq;
            float bv = bias[n];
#pragma unroll
            for (int rr = 0; rr < 8; ++rr) {
                int mm = (mt2 * 2 + mi) * 16 + rr + 8 * g;
                int bb = mm / op2;
                int r2 = mm % op2;
                float val = acc[mi][sub][rr] + bv;
                val = val > 0.f ? val : 0.f;
                Pout[(unsigned)bb * PSTRIDE + (unsigned)r2 * N + n] = f2bf(val);  // NHWC store
            }
        }
    }
}

// ----------------------------- WMMA FC GEMM ---------------------------------
// out = act( A(bf16 [256][K], stride lda) @ W^T + bias ).  PACKED: weights in
// fragment layout (natural k order); else raw fp32 [N][K] converted in-flight
// with a global_prefetch_b8 running ahead of the HBM weight stream (fc6).
template <bool PACKED>
__global__ __launch_bounds__(256)
void fc_gemm_wmma(const unsigned short* __restrict__ A, int lda,
                  const float* __restrict__ W, const unsigned int* __restrict__ Wpk,
                  const float* __restrict__ bias,
                  int K, int KT, int Nt64, int Nvalid,
                  unsigned short* __restrict__ outBf, float* __restrict__ outF,
                  int ldo, int relu) {
    int wid  = (int)((blockIdx.x * blockDim.x + threadIdx.x) >> 5);
    int lane = threadIdx.x & 31;
    if (wid >= 8 * Nt64) return;                  // M=256 -> 8 pairs of 16-row tiles
    int mt2 = wid & 7, nt = wid >> 3;
    int g = lane >> 4, lq = lane & 15;

    unsigned Arow[2];
#pragma unroll
    for (int mi = 0; mi < 2; ++mi) {
        int m = (mt2 * 2 + mi) * 16 + lq;
        Arow[mi] = (unsigned)m * (unsigned)lda;
    }
    const unsigned int* Bp[4];
#pragma unroll
    for (int sub = 0; sub < 4; ++sub) {
        int n16 = nt * 4 + sub;
        Bp[sub] = PACKED ? (Wpk + (size_t)n16 * KT * 256 + lane * 8) : nullptr;
    }

    v8f acc[2][4];
#pragma unroll
    for (int mi = 0; mi < 2; ++mi)
#pragma unroll
        for (int s2 = 0; s2 < 4; ++s2)
#pragma unroll
            for (int e = 0; e < 8; ++e) acc[mi][s2][e] = 0.f;

    union AFrag { v4u q[2]; v16bf bf; };

    for (int kt = 0; kt < KT; ++kt) {             // K is a multiple of 32
        int kb = kt * 32;
        v16bf a[2];
#pragma unroll
        for (int mi = 0; mi < 2; ++mi) {
            AFrag af;                              // runs k=kb+8g..+7, kb+16+8g..+7
            af.q[0] = *(const v4u*)(A + Arow[mi] + kb + 8 * g);
            af.q[1] = *(const v4u*)(A + Arow[mi] + kb + 16 + 8 * g);
            a[mi] = af.bf;
        }
#pragma unroll
        for (int sub = 0; sub < 4; ++sub) {
            v8u bu;
            if (PACKED) {
                bu = *(const v8u*)(Bp[sub] + (size_t)kt * 256);
            } else {
                int n = nt * 64 + sub * 16 + lq;
                const float* Wr = W + (size_t)n * K;
                if (kb + 512 < K)                  // per-lane stream prefetch, ~2KB ahead
                    __builtin_prefetch((const void*)(Wr + kb + 512), 0, 1);
#pragma unroll
                for (int v = 0; v < 8; ++v) {
                    int k0 = kb + ((v < 4) ? (8 * g + 2 * v) : (16 + 8 * g + 2 * (v - 4)));
                    bu[v] = pk2(f2bf(Wr[k0]), f2bf(Wr[k0 + 1]));
                }
            }
            v16bf bfrag = __builtin_bit_cast(v16bf, bu);
#pragma unroll
            for (int mi = 0; mi < 2; ++mi)
                acc[mi][sub] = __builtin_amdgcn_wmma_f32_16x16x32_bf16(
                    false, a[mi], false, bfrag, (short)0, acc[mi][sub], false, false);
        }
    }

#pragma unroll
    for (int mi = 0; mi < 2; ++mi) {
#pragma unroll
        for (int sub = 0; sub < 4; ++sub) {
            int n = nt * 64 + sub * 16 + lq;
            float bv = (n < Nvalid) ? bias[n] : 0.f;
#pragma unroll
            for (int rr = 0; rr < 8; ++rr) {
                int mm = (mt2 * 2 + mi) * 16 + rr + 8 * g;
                float val = acc[mi][sub][rr] + bv;
                if (relu) {
                    val = val > 0.f ? val : 0.f;
                    outBf[(size_t)mm * ldo + n] = f2bf(val);
                } else if (n < Nvalid) {
                    outF[(size_t)mm * ldo + n] = val;
                }
            }
        }
    }
}

__global__ void softmax1000(const float* __restrict__ logits, float* __restrict__ out) {
    int row = blockIdx.x, t = threadIdx.x;
    __shared__ float red[256];
    const float* lr = logits + (size_t)row * 1000;
    float mx = -3.4e38f;
    for (int i = t; i < 1000; i += 256) mx = fmaxf(mx, lr[i]);
    red[t] = mx; __syncthreads();
    for (int s = 128; s > 0; s >>= 1) { if (t < s) red[t] = fmaxf(red[t], red[t + s]); __syncthreads(); }
    mx = red[0]; __syncthreads();
    float sum = 0.f;
    for (int i = t; i < 1000; i += 256) sum += expf(lr[i] - mx);
    red[t] = sum; __syncthreads();
    for (int s = 128; s > 0; s >>= 1) { if (t < s) red[t] += red[t + s]; __syncthreads(); }
    float inv = 1.0f / red[0];
    for (int i = t; i < 1000; i += 256) out[(size_t)row * 1000 + i] = expf(lr[i] - mx) * inv;
}

// ------------------------------- launcher -----------------------------------

extern "C" void kernel_launch(void* const* d_in, const int* in_sizes, int n_in,
                              void* d_out, int out_size, void* d_ws, size_t ws_size,
                              hipStream_t stream) {
    (void)in_sizes; (void)n_in; (void)out_size; (void)ws_size;

    const float* image = (const float*)d_in[0];
    const float* patch = (const float*)d_in[1];
    const int*   locs0 = (const int*)d_in[2];
    const float* cw[13]; const float* cb[13];
    for (int i = 0; i < 13; ++i) { cw[i] = (const float*)d_in[4 + 2 * i]; cb[i] = (const float*)d_in[5 + 2 * i]; }
    const float* fc6w = (const float*)d_in[30]; const float* fc6b = (const float*)d_in[31];
    const float* fc7w = (const float*)d_in[32]; const float* fc7b = (const float*)d_in[33];
    const float* fc8w = (const float*)d_in[34]; const float* fc8b = (const float*)d_in[35];

    char* ws = (char*)d_ws;
    size_t off = 0;
    auto carve = [&](size_t bytes) -> void* {
        void* p = ws + off;
        off = (off + bytes + 255) & ~((size_t)255);
        return p;
    };

    // cached activations acts[0..17] (acts[0] = image input, read-only)
    static const int actC[18] = {3,64,64,64,128,128,128,256,256,256,256,512,512,512,512,512,512,512};
    static const int actH[18] = {224,224,224,112,112,112,56,56,56,56,28,28,28,28,14,14,14,14};
    const float* actP[18];
    float*       actW[18];
    actP[0] = image; actW[0] = nullptr;
    for (int i = 1; i < 18; ++i) {
        actW[i] = (float*)carve((size_t)actC[i] * actH[i] * actH[i] * sizeof(float));
        actP[i] = actW[i];
    }
    int* locsBuf[2] = {(int*)carve(256 * 2 * sizeof(int)), (int*)carve(256 * 2 * sizeof(int))};
    unsigned short* PA = (unsigned short*)carve((size_t)256 * PSTRIDE * 2);
    unsigned short* PB = (unsigned short*)carve((size_t)256 * PSTRIDE * 2);
    unsigned short* XB = (unsigned short*)carve((size_t)256 * PSTRIDE * 2);
    unsigned short* h6 = (unsigned short*)carve((size_t)256 * 4096 * 2);
    unsigned short* h7 = (unsigned short*)carve((size_t)256 * 4096 * 2);
    float* logits      = (float*)carve((size_t)256 * 1000 * sizeof(float));
    // packed-weight scratch: sized for fc7 (256 n16-tiles * 128 kt * 256 dwords)
    unsigned int* Wpk  = (unsigned int*)carve((size_t)256 * 128 * 256 * sizeof(unsigned int));

    // ---- full-image forward, layers 0..16 (pool5 itself is never needed) ----
    struct FullL { int kind, ci, C, H, O; };
    static const FullL FL[17] = {
        {0,0,3,224,64},{0,1,64,224,64},{1,-1,64,224,0},
        {0,2,64,112,128},{0,3,128,112,128},{1,-1,128,112,0},
        {0,4,128,56,256},{0,5,256,56,256},{0,6,256,56,256},{1,-1,256,56,0},
        {0,7,256,28,512},{0,8,512,28,512},{0,9,512,28,512},{1,-1,512,28,0},
        {0,10,512,14,512},{0,11,512,14,512},{0,12,512,14,512}};
    for (int i = 0; i < 17; ++i) {
        if (FL[i].kind == 0) {
            long long tot = (long long)FL[i].O * FL[i].H * FL[i].H;
            conv_full<<<(unsigned)((tot + 255) / 256), 256, 0, stream>>>(
                actP[i], cw[FL[i].ci], cb[FL[i].ci], actW[i + 1], FL[i].C, FL[i].H, FL[i].O);
        } else {
            int Ho = FL[i].H / 2;
            long long tot = (long long)FL[i].C * Ho * Ho;
            pool_full<<<(unsigned)((tot + 255) / 256), 256, 0, stream>>>(
                actP[i], actW[i + 1], FL[i].C, FL[i].H, Ho);
        }
    }

    // ---- incremental patch pipeline ----
    {
        long long tot = 256LL * 3 * 16 * 16;
        init_patch<<<(unsigned)((tot + 255) / 256), 256, 0, stream>>>(patch, PA);
    }
    struct PL { int kind, ci, C, szOut, szIn, s, p, k, out_p, in_p, prev, N; };
    static const PL PC[18] = {
        {0, 0,  3,224,224,1,1,3,18,20,16, 64},
        {0, 1, 64,224,224,1,1,3,20,22,18, 64},
        {1,-1, 64,112,224,2,0,2,11,22,20,  0},
        {0, 2, 64,112,112,1,1,3,13,15,11,128},
        {0, 3,128,112,112,1,1,3,15,17,13,128},
        {1,-1,128, 56,112,2,0,2, 8,16,15,  0},
        {0, 4,128, 56, 56,1,1,3,10,12, 8,256},
        {0, 5,256, 56, 56,1,1,3,12,14,10,256},
        {0, 6,256, 56, 56,1,1,3,14,16,12,256},
        {1,-1,256, 28, 56,2,0,2, 8,16,14,  0},
        {0, 7,256, 28, 28,1,1,3,10,12, 8,512},
        {0, 8,512, 28, 28,1,1,3,12,14,10,512},
        {0, 9,512, 28, 28,1,1,3,14,16,12,512},
        {1,-1,512, 14, 28,2,0,2, 8,16,14,  0},
        {0,10,512, 14, 14,1,1,3,10,12, 8,512},
        {0,11,512, 14, 14,1,1,3,12,14,10,512},
        {0,12,512, 14, 14,1,1,3,14,16,12,512},
        {1,-1,512,  7, 14,2,0,2, 7,14,14,  0}};

    const int* li = locs0;
    int lsel = 0;
    unsigned short* cur = PA;
    unsigned short* nxt = PB;
    for (int L = 0; L < 18; ++L) {
        const PL& c = PC[L];
        int* lo = locsBuf[lsel]; lsel ^= 1;
        update_locs<<<1, 256, 0, stream>>>(li, lo, c.s, c.p, c.k, c.out_p, c.szOut);
        long long tot = 256LL * c.C * c.in_p * c.in_p;
        assemble_patch<<<(unsigned)((tot + 255) / 256), 256, 0, stream>>>(
            actP[L], c.szIn, cur, c.prev, li, lo, XB, c.C, c.in_p, c.s, c.p, tot);
        if (c.kind == 0) {
            int K  = c.C * 9;
            int KT = (K + 31) / 32;
            long long ptot = (long long)(c.N / 16) * KT * 256;
            pack_weights<<<(unsigned)((ptot + 255) / 256), 256, 0, stream>>>(
                cw[c.ci], Wpk, K, c.N, KT, c.C, ptot);
            long long waves = (long long)(256 * c.out_p * c.out_p / 32) * (c.N / 64);
            if (K % 32 == 0)
                conv_patch_wmma<false><<<(unsigned)((waves * 32 + 255) / 256), 256, 0, stream>>>(
                    XB, Wpk, cb[c.ci], nxt, c.C, c.in_p, c.out_p, c.N, KT);
            else
                conv_patch_wmma<true><<<(unsigned)((waves * 32 + 255) / 256), 256, 0, stream>>>(
                    XB, Wpk, cb[c.ci], nxt, c.C, c.in_p, c.out_p, c.N, KT);
        } else {
            long long tot2 = 256LL * c.C * c.out_p * c.out_p;
            pool_patch<<<(unsigned)((tot2 + 255) / 256), 256, 0, stream>>>(
                XB, nxt, c.C, c.in_p, c.out_p, tot2);
        }
        unsigned short* t = cur; cur = nxt; nxt = t;
        li = lo;
    }
    // cur: pool5-equivalent NHWC patches [256][7][7][512], row stride PSTRIDE

    // ---- FC head (WMMA GEMMs) + softmax ----
    {   // NHWC -> NCHW flatten (reference reshape order) into the free X buffer
        long long tot = 256LL * 25088;
        pool5_transpose<<<(unsigned)((tot + 255) / 256), 256, 0, stream>>>(cur, XB);
    }
    // fc6: [256,25088] x [4096,25088]^T, relu -> bf16 (weights unpacked, cvt in-flight)
    fc_gemm_wmma<false><<<(8 * 64 * 32) / 256, 256, 0, stream>>>(
        XB, 25088, fc6w, nullptr, fc6b, 25088, 784, 64, 4096, h6, nullptr, 4096, 1);
    // fc7: [256,4096] x [4096,4096]^T, relu -> bf16 (packed weights)
    {
        long long ptot = 256LL * 128 * 256;
        pack_weights<<<(unsigned)((ptot + 255) / 256), 256, 0, stream>>>(
            fc7w, Wpk, 4096, 4096, 128, 0, ptot);
        fc_gemm_wmma<true><<<(8 * 64 * 32) / 256, 256, 0, stream>>>(
            h6, 4096, nullptr, Wpk, fc7b, 4096, 128, 64, 4096, h7, nullptr, 4096, 1);
    }
    // fc8: [256,4096] x [1000,4096]^T -> fp32 logits (N padded to 1024, guarded)
    {
        long long ptot = 64LL * 128 * 256;
        pack_weights<<<(unsigned)((ptot + 255) / 256), 256, 0, stream>>>(
            fc8w, Wpk, 4096, 1000, 128, 0, ptot);
        fc_gemm_wmma<true><<<(8 * 16 * 32) / 256, 256, 0, stream>>>(
            h7, 4096, nullptr, Wpk, fc8b, 4096, 128, 16, 1000, nullptr, logits, 1000, 0);
    }
    softmax1000<<<256, 256, 0, stream>>>(logits, (float*)d_out);
}
